// ZAIEModel_60687887892953
// MI455X (gfx1250) — matrix operand order repaired and checked by reading, not compiled, not run on hardware
//
#include <hip/hip_runtime.h>
#include <cstdint>
#include <cstddef>

// ---------------------------------------------------------------------------
// Problem constants (from the reference)
// ---------------------------------------------------------------------------
constexpr int V  = 50257;
constexpr int C  = 512;
constexpr int H  = 1024;
constexpr int E  = 8;
constexpr int NH = 16;
constexpr int Bb = 2;
constexpr int S  = 2048;
constexpr int LC = 128;
constexpr int HD = 64;          // H / NH
constexpr int MT = Bb * S;      // 4096 tokens
constexpr int ML = Bb * LC;     // 256 latent rows

// ---------------------------------------------------------------------------
// CDNA5 WMMA types (wave32, 16x16x32 bf16 -> f32)
// ---------------------------------------------------------------------------
typedef __attribute__((ext_vector_type(16))) __bf16 bf16x16;
typedef __attribute__((ext_vector_type(8)))  __bf16 bf16x8;
typedef __attribute__((ext_vector_type(8)))  float  f32x8;

static __device__ __forceinline__ int imin(int a, int b) { return a < b ? a : b; }

// ---------------------------------------------------------------------------
// fp32 -> bf16 cast (grid stride)
// ---------------------------------------------------------------------------
__global__ void k_cast_bf16(const float* __restrict__ in, __bf16* __restrict__ out, long n) {
    long i = (long)blockIdx.x * blockDim.x + threadIdx.x;
    long stride = (long)gridDim.x * blockDim.x;
    for (; i < n; i += stride) out[i] = (__bf16)in[i];
}

// ---------------------------------------------------------------------------
// Embedding gather + cast: te[m][d] = bf16(emb[tok[m]][d])
// ---------------------------------------------------------------------------
__global__ void k_gather_cast(const int* __restrict__ tok, const float* __restrict__ emb,
                              __bf16* __restrict__ te, int rows) {
    long i = (long)blockIdx.x * blockDim.x + threadIdx.x;
    long n = (long)rows * H;
    if (i >= n) return;
    long m = i >> 10;           // / H
    int  d = (int)(i & (H - 1));
    te[i] = (__bf16)emb[(long)tok[m] * H + d];
}

// ---------------------------------------------------------------------------
// Fragment loaders (contiguous-K wave32 lane layouts, same order for A and B).
//   A 16x32 bf16: lane l -> row m=(l&15), K-half (l>>4); elems = K kh*16..+15
//                 -> two 16B loads.
//   B 32x16 bf16: lane l -> col n=(l&15), K-half (l>>4); elems = K kh*16..+15
//                 contiguous in the (N,K) weight row -> one 32B load.
// ---------------------------------------------------------------------------
static __device__ __forceinline__ bf16x16 load_a_frag(const __bf16* p /* = row base + kh*16 */,
                                                      int k) {
    const bf16x8* ap = (const bf16x8*)(p + k);
    return __builtin_shufflevector(ap[0], ap[1],
                                   0,1,2,3,4,5,6,7,8,9,10,11,12,13,14,15);
}
static __device__ __forceinline__ bf16x16 load_b_frag(const __bf16* p /* = row base + kh*16 */,
                                                      int k) {
    return *(const bf16x16*)(p + k);
}

// ---------------------------------------------------------------------------
// Generic WMMA GEMM:  out[M,N] = A[M,K] * W[N,K]^T + bias[N]
// block = 256 threads = 8 waves (4 along M x 2 along N), wave tile 32x32,
// block tile 128x64. K-loop unrolled x2 (64 deep). Optional bf16 2nd output.
// ---------------------------------------------------------------------------
__global__ __launch_bounds__(256)
void k_gemm_wmma(const __bf16* __restrict__ A, const __bf16* __restrict__ W,
                 const float* __restrict__ bias,
                 float* __restrict__ outf, __bf16* __restrict__ outb,
                 int M, int N, int K) {
    const int lane = threadIdx.x & 31;
    const int wid  = threadIdx.x >> 5;
    const int wm   = wid & 3;
    const int wn   = wid >> 2;
    const int m0   = blockIdx.y * 128 + wm * 32;
    const int n0   = blockIdx.x * 64  + wn * 32;
    const int l15  = lane & 15;
    const int kh   = lane >> 4;     // K-half select

    // loop-invariant fragment base pointers
    const __bf16* aptr[2];
    const __bf16* bptr[2];
#pragma unroll
    for (int mt = 0; mt < 2; ++mt) {
        int m = imin(m0 + mt * 16 + l15, M - 1);
        aptr[mt] = A + (size_t)m * K + kh * 16;
    }
#pragma unroll
    for (int nt = 0; nt < 2; ++nt) {
        int n = imin(n0 + nt * 16 + l15, N - 1);
        bptr[nt] = W + (size_t)n * K + kh * 16;
    }

    f32x8 acc[2][2] = {};

    for (int k0 = 0; k0 < K; k0 += 64) {
        // prefetch streaming B operand one chunk ahead
        __builtin_prefetch(bptr[0] + k0 + 64, 0, 1);
        __builtin_prefetch(bptr[1] + k0 + 64, 0, 1);
#pragma unroll
        for (int ks = 0; ks < 2; ++ks) {
            const int k = k0 + ks * 32;
            bf16x16 a[2], b[2];
#pragma unroll
            for (int mt = 0; mt < 2; ++mt) a[mt] = load_a_frag(aptr[mt], k);
#pragma unroll
            for (int nt = 0; nt < 2; ++nt) b[nt] = load_b_frag(bptr[nt], k);
#pragma unroll
            for (int mt = 0; mt < 2; ++mt)
#pragma unroll
                for (int nt = 0; nt < 2; ++nt)
                    acc[mt][nt] = __builtin_amdgcn_wmma_f32_16x16x32_bf16(
                        false, a[mt], false, b[nt], (short)0, acc[mt][nt], false, false);
        }
    }

    const bool hasF = (outf != nullptr);
    const bool hasB = (outb != nullptr);
#pragma unroll
    for (int mt = 0; mt < 2; ++mt) {
#pragma unroll
        for (int nt = 0; nt < 2; ++nt) {
            int n = n0 + nt * 16 + l15;
            if (n >= N) continue;
            float bv = bias ? bias[n] : 0.0f;
#pragma unroll
            for (int r = 0; r < 8; ++r) {
                int m = m0 + mt * 16 + r + 8 * kh;
                if (m >= M) continue;
                float val = acc[mt][nt][r] + bv;
                if (hasF) outf[(size_t)m * N + n] = val;
                if (hasB) outb[(size_t)m * N + n] = (__bf16)val;
            }
        }
    }
}

// ---------------------------------------------------------------------------
// MoE WMMA GEMM:
//   moe[m,n] = sum_e gate[m,e]*(fused[m,:] . expert_w[e][n,:]) + sum_e gate[m,e]*eb[e,n]
// Same tiling; expert loop outside the K loop, per-row gate scaling on the
// expert accumulator before folding into the final accumulator.
// ---------------------------------------------------------------------------
__global__ __launch_bounds__(256)
void k_moe_wmma(const __bf16* __restrict__ A,    // fused bf16 (M x K)
                const __bf16* __restrict__ Wex,  // (E x N x K)
                const float*  __restrict__ gate, // (M x E)
                const float*  __restrict__ eb,   // (E x N)
                __bf16* __restrict__ outb,       // (M x N)
                int M, int N, int K) {
    const int lane = threadIdx.x & 31;
    const int wid  = threadIdx.x >> 5;
    const int wm   = wid & 3;
    const int wn   = wid >> 2;
    const int m0   = blockIdx.y * 128 + wm * 32;
    const int n0   = blockIdx.x * 64  + wn * 32;
    const int l15  = lane & 15;
    const int kh   = lane >> 4;

    const __bf16* aptr[2];
#pragma unroll
    for (int mt = 0; mt < 2; ++mt) {
        int m = imin(m0 + mt * 16 + l15, M - 1);
        aptr[mt] = A + (size_t)m * K + kh * 16;
    }
    const size_t brow[2] = {
        (size_t)imin(n0 + 0 * 16 + l15, N - 1) * K + (size_t)(kh * 16),
        (size_t)imin(n0 + 1 * 16 + l15, N - 1) * K + (size_t)(kh * 16)
    };

    f32x8 accF[2][2] = {};

    for (int e = 0; e < E; ++e) {
        const __bf16* We = Wex + (size_t)e * N * K;
        const __bf16* bptr[2] = { We + brow[0], We + brow[1] };
        f32x8 acc[2][2] = {};
        for (int k0 = 0; k0 < K; k0 += 64) {
            __builtin_prefetch(bptr[0] + k0 + 64, 0, 1);
            __builtin_prefetch(bptr[1] + k0 + 64, 0, 1);
#pragma unroll
            for (int ks = 0; ks < 2; ++ks) {
                const int k = k0 + ks * 32;
                bf16x16 a[2], b[2];
#pragma unroll
                for (int mt = 0; mt < 2; ++mt) a[mt] = load_a_frag(aptr[mt], k);
#pragma unroll
                for (int nt = 0; nt < 2; ++nt) b[nt] = load_b_frag(bptr[nt], k);
#pragma unroll
                for (int mt = 0; mt < 2; ++mt)
#pragma unroll
                    for (int nt = 0; nt < 2; ++nt)
                        acc[mt][nt] = __builtin_amdgcn_wmma_f32_16x16x32_bf16(
                            false, a[mt], false, b[nt], (short)0, acc[mt][nt], false, false);
            }
        }
        // per-row gate scale, accumulate into final
        float gv[2][8];
#pragma unroll
        for (int mt = 0; mt < 2; ++mt)
#pragma unroll
            for (int r = 0; r < 8; ++r) {
                int m = imin(m0 + mt * 16 + r + 8 * kh, M - 1);
                gv[mt][r] = gate[(size_t)m * E + e];
            }
#pragma unroll
        for (int mt = 0; mt < 2; ++mt)
#pragma unroll
            for (int nt = 0; nt < 2; ++nt)
#pragma unroll
                for (int r = 0; r < 8; ++r)
                    accF[mt][nt][r] += gv[mt][r] * acc[mt][nt][r];
    }

#pragma unroll
    for (int mt = 0; mt < 2; ++mt) {
#pragma unroll
        for (int nt = 0; nt < 2; ++nt) {
            int n = n0 + nt * 16 + l15;
            if (n >= N) continue;
#pragma unroll
            for (int r = 0; r < 8; ++r) {
                int m = m0 + mt * 16 + r + 8 * kh;
                if (m >= M) continue;
                float val = accF[mt][nt][r];
#pragma unroll
                for (int e = 0; e < E; ++e)
                    val += gate[(size_t)m * E + e] * eb[(size_t)e * N + n];
                outb[(size_t)m * N + n] = (__bf16)val;
            }
        }
    }
}

// ---------------------------------------------------------------------------
// Attention: one block per (b,s) token, 16 waves = 16 heads.
// scores (LC=128 keys) -> softmax -> ctx; head-mean of probs -> attn_weights.
// q: (B,S,H) f32, kv: (B,LC,2H) f32 (k at 0, v at +H).
// ---------------------------------------------------------------------------
__global__ __launch_bounds__(512)
void k_attention(const float* __restrict__ q, const float* __restrict__ kv,
                 __bf16* __restrict__ ctx, float* __restrict__ attnw) {
    __shared__ float qs[H];
    __shared__ float ps[NH * LC];

    const int bs   = blockIdx.x;        // b*S + s
    const int b    = bs / S;
    const int tid  = threadIdx.x;
    const int h    = tid >> 5;
    const int lane = tid & 31;

    for (int i = tid; i < H; i += 512) qs[i] = q[(size_t)bs * H + i];
    __syncthreads();

    const float scale = 0.125f;         // 1/sqrt(64)
    float sc[4];
#pragma unroll
    for (int jj = 0; jj < 4; ++jj) {
        int j = lane + jj * 32;
        const float* kp = kv + ((size_t)(b * LC + j) * (2 * H) + h * HD);
        float acc = 0.0f;
#pragma unroll 8
        for (int d = 0; d < HD; ++d) acc += qs[h * HD + d] * kp[d];
        sc[jj] = acc * scale;
    }
    float mx = fmaxf(fmaxf(sc[0], sc[1]), fmaxf(sc[2], sc[3]));
#pragma unroll
    for (int o = 16; o > 0; o >>= 1) mx = fmaxf(mx, __shfl_xor(mx, o, 32));
    float sum = 0.0f;
#pragma unroll
    for (int jj = 0; jj < 4; ++jj) { sc[jj] = __expf(sc[jj] - mx); sum += sc[jj]; }
#pragma unroll
    for (int o = 16; o > 0; o >>= 1) sum += __shfl_xor(sum, o, 32);
    float inv = 1.0f / sum;
#pragma unroll
    for (int jj = 0; jj < 4; ++jj) ps[h * LC + lane + jj * 32] = sc[jj] * inv;
    __syncthreads();

    if (tid < LC) {
        float s = 0.0f;
#pragma unroll
        for (int hh = 0; hh < NH; ++hh) s += ps[hh * LC + tid];
        attnw[(size_t)bs * LC + tid] = s * (1.0f / NH);
    }

#pragma unroll
    for (int dd = 0; dd < 2; ++dd) {
        int d = lane + dd * 32;
        float acc = 0.0f;
        for (int j = 0; j < LC; ++j)
            acc += ps[h * LC + j] * kv[(size_t)(b * LC + j) * (2 * H) + H + h * HD + d];
        ctx[(size_t)bs * H + h * HD + d] = (__bf16)acc;
    }
}

// ---------------------------------------------------------------------------
// Gate: softmax(fused @ planner_w.T + planner_b) per token. One wave/token.
// ---------------------------------------------------------------------------
__global__ __launch_bounds__(256)
void k_gate(const float* __restrict__ fused, const float* __restrict__ pw,
            const float* __restrict__ pb, float* __restrict__ gate, int M) {
    const int wid  = threadIdx.x >> 5;
    const int lane = threadIdx.x & 31;
    const int t    = blockIdx.x * 8 + wid;
    if (t >= M) return;

    float acc[E] = {};
    const float* fr = fused + (size_t)t * H;
    for (int d = lane; d < H; d += 32) {
        float f = fr[d];
#pragma unroll
        for (int e = 0; e < E; ++e) acc[e] += f * pw[e * H + d];
    }
#pragma unroll
    for (int e = 0; e < E; ++e)
#pragma unroll
        for (int o = 16; o > 0; o >>= 1) acc[e] += __shfl_xor(acc[e], o, 32);

    float mx = -1e30f;
#pragma unroll
    for (int e = 0; e < E; ++e) { acc[e] += pb[e]; mx = fmaxf(mx, acc[e]); }
    float sum = 0.0f;
#pragma unroll
    for (int e = 0; e < E; ++e) { acc[e] = __expf(acc[e] - mx); sum += acc[e]; }
    float inv = 1.0f / sum;
#pragma unroll
    for (int e = 0; e < E; ++e)
        if (lane == e) gate[(size_t)t * E + e] = acc[e] * inv;
}

// ---------------------------------------------------------------------------
// Host-side launcher
// ---------------------------------------------------------------------------
static inline int cdiv(int a, int b) { return (a + b - 1) / b; }

extern "C" void kernel_launch(void* const* d_in, const int* in_sizes, int n_in,
                              void* d_out, int out_size, void* d_ws, size_t ws_size,
                              hipStream_t stream) {
    (void)in_sizes; (void)n_in; (void)out_size; (void)ws_size;

    const int*   text = (const int*)  d_in[0];
    const float* ci   = (const float*)d_in[1];
    const float* emb  = (const float*)d_in[2];
    const float* encw = (const float*)d_in[3];
    const float* encb = (const float*)d_in[4];
    const float* ipw  = (const float*)d_in[5];
    const float* ipb  = (const float*)d_in[6];
    const float* aow  = (const float*)d_in[7];
    const float* aob  = (const float*)d_in[8];
    const float* plw  = (const float*)d_in[9];
    const float* plb  = (const float*)d_in[10];
    const float* exw  = (const float*)d_in[11];
    const float* exb  = (const float*)d_in[12];
    const float* ow   = (const float*)d_in[13];
    const float* ob   = (const float*)d_in[14];

    float* logits = (float*)d_out;
    float* attnw  = (float*)d_out + (size_t)MT * V;

    // ---- workspace carve-up (all sizes are multiples of 256 B) ----
    char* ws = (char*)d_ws;
    size_t off = 0;
    auto alloc = [&](size_t bytes) { char* p = ws + off; off += bytes; return p; };

    __bf16* te_b    = (__bf16*)alloc((size_t)MT * H * 2);       // 8 MB
    __bf16* encw_b  = (__bf16*)alloc((size_t)H * C * 2);        // 1 MB
    __bf16* ipw_b   = (__bf16*)alloc((size_t)3 * H * H * 2);    // 6 MB
    __bf16* aow_b   = (__bf16*)alloc((size_t)H * H * 2);        // 2 MB
    __bf16* exw_b   = (__bf16*)alloc((size_t)E * H * H * 2);    // 16 MB
    __bf16* ow_b    = (__bf16*)alloc((size_t)V * H * 2);        // 103 MB
    __bf16* ci_b    = (__bf16*)alloc((size_t)ML * C * 2);       // 0.25 MB
    float*  ce_f    = (float*) alloc((size_t)ML * H * 4);       // 1 MB
    __bf16* ce_b    = (__bf16*)alloc((size_t)ML * H * 2);       // 0.5 MB
    float*  kv_f    = (float*) alloc((size_t)ML * 2 * H * 4);   // 2 MB
    float*  q_f     = (float*) alloc((size_t)MT * H * 4);       // 16 MB
    __bf16* ctx_b   = (__bf16*)alloc((size_t)MT * H * 2);       // 8 MB
    float*  fused_f = (float*) alloc((size_t)MT * H * 4);       // 16 MB
    __bf16* fused_b = (__bf16*)alloc((size_t)MT * H * 2);       // 8 MB
    float*  gate_f  = (float*) alloc((size_t)MT * E * 4);       // 128 KB
    __bf16* moe_b   = (__bf16*)alloc((size_t)MT * H * 2);       // 8 MB

    auto cast = [&](const float* src, __bf16* dst, long n) {
        int blocks = (int)((n + 255) / 256);
        if (blocks > 65535 * 8) blocks = 65535 * 8;
        hipLaunchKernelGGL(k_cast_bf16, dim3(blocks), dim3(256), 0, stream, src, dst, n);
    };

    // 1) bf16 conversions
    cast(encw, encw_b, (long)H * C);
    cast(ipw,  ipw_b,  (long)3 * H * H);
    cast(aow,  aow_b,  (long)H * H);
    cast(exw,  exw_b,  (long)E * H * H);
    cast(ow,   ow_b,   (long)V * H);
    cast(ci,   ci_b,   (long)ML * C);

    // 2) embedding gather -> bf16
    {
        long n = (long)MT * H;
        hipLaunchKernelGGL(k_gather_cast, dim3((int)((n + 255) / 256)), dim3(256), 0, stream,
                           text, emb, te_b, MT);
    }

    // 3) ce = conceptual @ enc_w^T + enc_b   (M=256, N=1024, K=512)
    hipLaunchKernelGGL(k_gemm_wmma, dim3(cdiv(H, 64), cdiv(ML, 128)), dim3(256), 0, stream,
                       ci_b, encw_b, encb, ce_f, ce_b, ML, H, C);

    // 4) q = te @ wq^T + bq                  (M=4096, N=1024, K=1024)
    hipLaunchKernelGGL(k_gemm_wmma, dim3(cdiv(H, 64), cdiv(MT, 128)), dim3(256), 0, stream,
                       te_b, ipw_b, ipb, q_f, (__bf16*)nullptr, MT, H, H);

    // 5) kv = ce @ [wk;wv]^T + [bk;bv]       (M=256, N=2048, K=1024)
    hipLaunchKernelGGL(k_gemm_wmma, dim3(cdiv(2 * H, 64), cdiv(ML, 128)), dim3(256), 0, stream,
                       ce_b, ipw_b + (size_t)H * H, ipb + H, kv_f, (__bf16*)nullptr,
                       ML, 2 * H, H);

    // 6) attention -> ctx (bf16), attn_weights (fp32, second output)
    hipLaunchKernelGGL(k_attention, dim3(MT), dim3(512), 0, stream, q_f, kv_f, ctx_b, attnw);

    // 7) fused = ctx @ attn_out_w^T + b      (M=4096, N=1024, K=1024)
    hipLaunchKernelGGL(k_gemm_wmma, dim3(cdiv(H, 64), cdiv(MT, 128)), dim3(256), 0, stream,
                       ctx_b, aow_b, aob, fused_f, fused_b, MT, H, H);

    // 8) gate = softmax(fused @ planner^T + pb)
    hipLaunchKernelGGL(k_gate, dim3(cdiv(MT, 8)), dim3(256), 0, stream,
                       fused_f, plw, plb, gate_f, MT);

    // 9) moe (expert mixture, gate-weighted) -> bf16
    hipLaunchKernelGGL(k_moe_wmma, dim3(cdiv(H, 64), cdiv(MT, 128)), dim3(256), 0, stream,
                       fused_b, exw_b, gate_f, exb, moe_b, MT, H, H);

    // 10) logits = moe @ out_w^T + out_b     (M=4096, N=50257, K=1024)
    hipLaunchKernelGGL(k_gemm_wmma, dim3(cdiv(V, 64), cdiv(MT, 128)), dim3(256), 0, stream,
                       moe_b, ow_b, ob, logits, (__bf16*)nullptr, MT, V, H);
}
